// LongContextTransformerMoE_27075473834351
// MI455X (gfx1250) — compile-verified, matrix operand order checked by reading, not run on hardware
//
#include <hip/hip_runtime.h>
#include <math.h>
#include <stdint.h>

#define T_TOK 2048
#define DMODEL 1024
#define NHEAD 16
#define HDIM 64
#define NLAYER 3
#define DFF_ 4096
#define NEXP 8
#define SEQ 1024
#define BATCH 2
#define VOCAB 32000

typedef __attribute__((ext_vector_type(16))) __bf16 v16bf;
typedef __attribute__((ext_vector_type(8)))  __bf16 v8bf;
typedef __attribute__((ext_vector_type(2)))  __bf16 bf16x2;
typedef __attribute__((ext_vector_type(8)))  float  v8f;
typedef __attribute__((ext_vector_type(4)))  float  f32x4;

// ---------------------------------------------------------------------------
// WMMA fragment load from an LDS tile stored row-major [row][k] (k contiguous).
// CDNA5 16-bit A/B layout: lanes 0-15 -> row = lane, K in {0..7,16..23};
// lanes 16-31 -> row = lane-16, K in {8..15,24..31}.
// ---------------------------------------------------------------------------
__device__ __forceinline__ v16bf frag_ld(const __bf16* p, int ld, int row0, int k0) {
  const int lane = threadIdx.x & 31;
  const __bf16* q = p + (size_t)(row0 + (lane & 15)) * ld + k0;
  const int h = (lane >> 4) << 3;   // 0 or 8
  v16bf f;
#pragma unroll
  for (int e = 0; e < 8; ++e) f[e] = q[h + e];
#pragma unroll
  for (int e = 0; e < 8; ++e) f[8 + e] = q[16 + h + e];
  return f;
}

__device__ __forceinline__ v8f wmma_bf16(v16bf a, v16bf b, v8f c) {
  return __builtin_amdgcn_wmma_f32_16x16x32_bf16(false, a, false, b, (short)0, c,
                                                 false, false);
}

// Async global->LDS 16-byte copy (CDNA5 ASYNCcnt path).
__device__ __forceinline__ void async_b128(uint32_t lds_byte_addr,
                                           const void* gaddr) {
  asm volatile("global_load_async_to_lds_b128 %0, %1, off"
               :: "v"(lds_byte_addr), "v"((uint64_t)(uintptr_t)gaddr)
               : "memory");
}
__device__ __forceinline__ void wait_async0() {
  asm volatile("s_wait_asynccnt 0x0" ::: "memory");
}

// ---------------------------------------------------------------------------
// Generic GEMM: C[M,N] = f(A[M,K] @ W[K,N] + bias) (+residual) (*row_scale +=)
// 256 threads (8 waves), 128x64 C tile, K stepped by 32.
// fp32 in/out, bf16 WMMA compute; vectorized b128 staging.
// ---------------------------------------------------------------------------
__global__ void gemm_kernel(const float* __restrict__ A, const float* __restrict__ W,
                            const float* __restrict__ bias,
                            const float* __restrict__ residual,
                            const float* __restrict__ rscale, int rstride,
                            float* __restrict__ C,
                            int M, int Kd, int N, int do_relu, int accum) {
  __shared__ __bf16 As[128 * 32];  // [m][k]
  __shared__ __bf16 Bs[64 * 32];   // [n][k]  (K-transposed stage)
  const int tid  = threadIdx.x;
  const int lane = tid & 31;
  const int wave = tid >> 5;
  const int m0 = blockIdx.y * 128;
  const int n0 = blockIdx.x * 64;
  const int wm = (wave >> 1) * 32;   // 0,32,64,96
  const int wn = (wave & 1) * 32;    // 0,32
  (void)M;

  v8f vzero = {};
  v8f acc[2][2];
  for (int i = 0; i < 2; ++i)
    for (int j = 0; j < 2; ++j) acc[i][j] = vzero;

  // B staging task (fixed per thread): column n, 8-wide k chunk.
  const int bn = tid & 63;
  const int bk = (tid >> 6) << 3;    // 0,8,16,24

  for (int k0 = 0; k0 < Kd; k0 += 32) {
    // A tile 128x32: two tasks/thread, each = 2 x b128 load + packed b128 store.
#pragma unroll
    for (int it = 0; it < 2; ++it) {
      const int i = tid + it * 256;
      const int m = i >> 2, kc = (i & 3) << 3;
      const float* ap = A + (size_t)(m0 + m) * Kd + (k0 + kc);
      const f32x4 f0 = *(const f32x4*)ap;
      const f32x4 f1 = *(const f32x4*)(ap + 4);
      v8bf h;
#pragma unroll
      for (int j = 0; j < 4; ++j) h[j] = (__bf16)f0[j];
#pragma unroll
      for (int j = 0; j < 4; ++j) h[4 + j] = (__bf16)f1[j];
      *(v8bf*)&As[m * 32 + kc] = h;
    }
    // B tile 32x64 -> transposed [n][k]: strided reads (coalesced across lanes),
    // one packed b128 LDS store per thread.
    {
      const float* wp = W + (size_t)(k0 + bk) * N + (n0 + bn);
      v8bf h;
#pragma unroll
      for (int j = 0; j < 8; ++j) h[j] = (__bf16)wp[(size_t)j * N];
      *(v8bf*)&Bs[bn * 32 + bk] = h;
    }
    __syncthreads();
    const v16bf a0 = frag_ld(As, 32, wm,      0);
    const v16bf a1 = frag_ld(As, 32, wm + 16, 0);
    const v16bf b0 = frag_ld(Bs, 32, wn,      0);
    const v16bf b1 = frag_ld(Bs, 32, wn + 16, 0);
    acc[0][0] = wmma_bf16(a0, b0, acc[0][0]);
    acc[0][1] = wmma_bf16(a0, b1, acc[0][1]);
    acc[1][0] = wmma_bf16(a1, b0, acc[1][0]);
    acc[1][1] = wmma_bf16(a1, b1, acc[1][1]);
    __syncthreads();
  }

  // Epilogue. C/D layout: VGPR r, lanes 0-15 -> M=r, lanes 16-31 -> M=r+8.
  const int rofs = (lane >> 4) << 3;
  const int cofs = lane & 15;
#pragma unroll
  for (int fi = 0; fi < 2; ++fi)
#pragma unroll
    for (int fj = 0; fj < 2; ++fj)
#pragma unroll
      for (int r = 0; r < 8; ++r) {
        const int row = m0 + wm + 16 * fi + r + rofs;
        const int col = n0 + wn + 16 * fj + cofs;
        float v = acc[fi][fj][r] + bias[col];
        if (do_relu) v = fmaxf(v, 0.0f);
        if (residual) v += residual[(size_t)row * N + col];
        const size_t oi = (size_t)row * N + col;
        if (accum) C[oi] += rscale[(size_t)row * rstride] * v;
        else       C[oi] = v;
      }
}

// ---------------------------------------------------------------------------
// Convert fp32 q,k,v -> bf16 (q pre-scaled by 1/sqrt(HD); v transposed to
// [b,h,d,s]) so attention tiles are contiguous 128B rows for async-LDS loads.
// ---------------------------------------------------------------------------
__global__ void cvt_qkv_kernel(const float* __restrict__ q, const float* __restrict__ k,
                               const float* __restrict__ v,
                               __bf16* __restrict__ qh, __bf16* __restrict__ kh,
                               __bf16* __restrict__ vt) {
  const int idx = blockIdx.x * 256 + threadIdx.x;
  if (idx >= T_TOK * DMODEL) return;
  qh[idx] = (__bf16)(q[idx] * 0.125f);
  kh[idx] = (__bf16)k[idx];
  const int t = idx >> 10;
  const int hd = idx & (DMODEL - 1);
  const int h = hd >> 6, d = hd & 63;
  const int b = t >> 10;            // t / SEQ
  const int s = t & (SEQ - 1);
  vt[((size_t)(b * NHEAD + h) * HDIM + d) * SEQ + s] = (__bf16)v[idx];
}

// ---------------------------------------------------------------------------
// Flash attention: one workgroup (128 thr / 4 waves) per (64-q-tile, head, b).
// All tile staging via GLOBAL_LOAD_ASYNC_TO_LDS_B128; QK^T and P.V on WMMA.
// ---------------------------------------------------------------------------
__global__ void attn_kernel(const __bf16* __restrict__ qh,
                            const __bf16* __restrict__ kh,
                            const __bf16* __restrict__ vt,
                            float* __restrict__ o) {
  __shared__ __bf16 Qs[64 * 64];   // [qrow][d]
  __shared__ __bf16 Ks[64 * 64];   // [kvrow][d]
  __shared__ __bf16 Vs[64 * 64];   // [d][kvrow] (pre-transposed in global)
  __shared__ __bf16 Ps[64 * 64];   // [qrow][kvrow]
  __shared__ float  Ss[64 * 64];
  __shared__ float  mrow[64], lrow[64], frow[64];

  const int tid  = threadIdx.x;
  const int lane = tid & 31;
  const int wave = tid >> 5;
  const int wm = (wave >> 1) * 32;
  const int wn = (wave & 1) * 32;
  const int t0 = blockIdx.x * 64;
  const int hh = blockIdx.y;
  const int b  = blockIdx.z;
  const size_t base = (size_t)b * SEQ;

  const uint32_t ldsQ = (uint32_t)(uintptr_t)&Qs[0];
  const uint32_t ldsK = (uint32_t)(uintptr_t)&Ks[0];
  const uint32_t ldsV = (uint32_t)(uintptr_t)&Vs[0];
  const __bf16* vrow = vt + (size_t)(b * NHEAD + hh) * HDIM * SEQ;

  // Q tile: 64 rows x 128B, async copied.
#pragma unroll
  for (int it = 0; it < 4; ++it) {
    const int i = tid + it * 128;
    const int r = i >> 3, seg = (i & 7) << 4;   // 16B segments
    async_b128(ldsQ + (r << 7) + seg,
               (const char*)(qh + (base + t0 + r) * DMODEL + hh * HDIM) + seg);
  }
  if (tid < 64) { mrow[tid] = -1e30f; lrow[tid] = 0.0f; }

  v8f vzero = {};
  v8f oacc[2][2];
  for (int i = 0; i < 2; ++i)
    for (int j = 0; j < 2; ++j) oacc[i][j] = vzero;

  const int rofs = (lane >> 4) << 3;
  const int cofs = lane & 15;

  for (int jt = 0; jt < SEQ / 64; ++jt) {
    __syncthreads();  // previous iteration's consumers done
#pragma unroll
    for (int it = 0; it < 4; ++it) {
      const int i = tid + it * 128;
      const int r = i >> 3, seg = (i & 7) << 4;
      async_b128(ldsK + (r << 7) + seg,
                 (const char*)(kh + (base + jt * 64 + r) * DMODEL + hh * HDIM) + seg);
      async_b128(ldsV + (r << 7) + seg,
                 (const char*)(vrow + (size_t)r * SEQ + jt * 64) + seg);
    }
    wait_async0();
    __syncthreads();

    // S = Q @ K^T  (Kdim = 64 -> two 32-wide WMMA steps)
    v8f sacc[2][2];
    for (int i = 0; i < 2; ++i)
      for (int j = 0; j < 2; ++j) sacc[i][j] = vzero;
#pragma unroll
    for (int kk = 0; kk < 2; ++kk) {
      const v16bf a0 = frag_ld(Qs, 64, wm,      kk * 32);
      const v16bf a1 = frag_ld(Qs, 64, wm + 16, kk * 32);
      const v16bf b0 = frag_ld(Ks, 64, wn,      kk * 32);
      const v16bf b1 = frag_ld(Ks, 64, wn + 16, kk * 32);
      sacc[0][0] = wmma_bf16(a0, b0, sacc[0][0]);
      sacc[0][1] = wmma_bf16(a0, b1, sacc[0][1]);
      sacc[1][0] = wmma_bf16(a1, b0, sacc[1][0]);
      sacc[1][1] = wmma_bf16(a1, b1, sacc[1][1]);
    }
#pragma unroll
    for (int fi = 0; fi < 2; ++fi)
#pragma unroll
      for (int fj = 0; fj < 2; ++fj)
#pragma unroll
        for (int r = 0; r < 8; ++r)
          Ss[(wm + 16 * fi + r + rofs) * 64 + (wn + 16 * fj + cofs)] =
              sacc[fi][fj][r];
    __syncthreads();

    // Online softmax: one thread per q row; packed bf16 P stores.
    if (tid < 64) {
      float mx = -1e30f;
      for (int c = 0; c < 64; ++c) mx = fmaxf(mx, Ss[tid * 64 + c]);
      const float mn = fmaxf(mrow[tid], mx);
      const float f  = __expf(mrow[tid] - mn);
      float s = 0.0f;
      for (int c = 0; c < 64; c += 2) {
        const float p0 = __expf(Ss[tid * 64 + c]     - mn);
        const float p1 = __expf(Ss[tid * 64 + c + 1] - mn);
        bf16x2 h; h[0] = (__bf16)p0; h[1] = (__bf16)p1;
        *(bf16x2*)&Ps[tid * 64 + c] = h;
        s += p0 + p1;
      }
      lrow[tid] = lrow[tid] * f + s;
      frow[tid] = f;
      mrow[tid] = mn;
    }
    __syncthreads();

    // Rescale O accumulator, then O += P @ V.
#pragma unroll
    for (int fi = 0; fi < 2; ++fi)
#pragma unroll
      for (int r = 0; r < 8; ++r) {
        const float f = frow[wm + 16 * fi + r + rofs];
        oacc[fi][0][r] *= f;
        oacc[fi][1][r] *= f;
      }
#pragma unroll
    for (int kk = 0; kk < 2; ++kk) {
      const v16bf a0 = frag_ld(Ps, 64, wm,      kk * 32);
      const v16bf a1 = frag_ld(Ps, 64, wm + 16, kk * 32);
      const v16bf b0 = frag_ld(Vs, 64, wn,      kk * 32);
      const v16bf b1 = frag_ld(Vs, 64, wn + 16, kk * 32);
      oacc[0][0] = wmma_bf16(a0, b0, oacc[0][0]);
      oacc[0][1] = wmma_bf16(a0, b1, oacc[0][1]);
      oacc[1][0] = wmma_bf16(a1, b0, oacc[1][0]);
      oacc[1][1] = wmma_bf16(a1, b1, oacc[1][1]);
    }
  }

#pragma unroll
  for (int fi = 0; fi < 2; ++fi)
#pragma unroll
    for (int fj = 0; fj < 2; ++fj)
#pragma unroll
      for (int r = 0; r < 8; ++r) {
        const int row = wm + 16 * fi + r + rofs;
        const int col = wn + 16 * fj + cofs;
        o[(base + t0 + row) * DMODEL + hh * HDIM + col] =
            oacc[fi][fj][r] / lrow[row];
      }
}

// ---------------------------------------------------------------------------
// YaRN cos/sin table: grid=SEQ, block=32.
// ---------------------------------------------------------------------------
__global__ void yarn_kernel(float* __restrict__ cosb, float* __restrict__ sinb) {
  const int s = blockIdx.x;
  const int i = threadIdx.x;     // 0..31
  const float exp_i = (float)i / 32.0f;
  const float fe = powf(10000.0f, -exp_i);
  const float fi_ = fe / 8.0f;
  const float pi2 = 6.2831853071795864f;
  const float denom = 2.0f * logf(10000.0f);
  const float cf = 64.0f * logf(2048.0f / (32.0f * pi2)) / denom;
  const float cs = 64.0f * logf(2048.0f / ( 1.0f * pi2)) / denom;
  const float low  = fmaxf(floorf(cf), 0.0f);
  const float high = fminf(ceilf(cs), 63.0f);
  const float lin  = ((float)i - low) / fmaxf(high - low, 1e-3f);
  const float mask = 1.0f - fminf(fmaxf(lin, 0.0f), 1.0f);
  const float invf = fi_ * (1.0f - mask) + fe * mask;
  const float ms = 0.1f * logf(8.0f) + 1.0f;
  const float ang = (float)s * invf;
  cosb[s * 32 + i] = cosf(ang) * ms;
  sinb[s * 32 + i] = sinf(ang) * ms;
}

// RoPE in place on q and k.
__global__ void rope_kernel(float* __restrict__ q, float* __restrict__ k,
                            const float* __restrict__ cosb,
                            const float* __restrict__ sinb) {
  const int idx = blockIdx.x * 256 + threadIdx.x;
  if (idx >= T_TOK * NHEAD * 32) return;
  const int dd = idx & 31;
  const int h  = (idx >> 5) & (NHEAD - 1);
  const int t  = idx >> 9;
  const int s  = t & (SEQ - 1);
  const float c  = cosb[s * 32 + dd];
  const float sn = sinb[s * 32 + dd];
  const size_t base = (size_t)t * DMODEL + h * HDIM + dd;
  const float q1 = q[base], q2 = q[base + 32];
  q[base]      = q1 * c - q2 * sn;
  q[base + 32] = q2 * c + q1 * sn;
  const float k1 = k[base], k2 = k[base + 32];
  k[base]      = k1 * c - k2 * sn;
  k[base + 32] = k2 * c + k1 * sn;
}

// Embedding gather: x = emb[clip(src)] * sqrt(D)
__global__ void embed_kernel(const int* __restrict__ src,
                             const float* __restrict__ emb,
                             float* __restrict__ x) {
  const int idx = blockIdx.x * 256 + threadIdx.x;
  if (idx >= T_TOK * DMODEL) return;
  const int t = idx >> 10;
  const int d = idx & (DMODEL - 1);
  int tok = src[t];
  tok = tok < 0 ? 0 : (tok >= VOCAB ? VOCAB - 1 : tok);
  x[idx] = emb[(size_t)tok * DMODEL + d] * 32.0f;   // sqrt(1024)
}

// LayerNorm over D per row.
__global__ void ln_kernel(const float* __restrict__ x, const float* __restrict__ g,
                          const float* __restrict__ bta, float* __restrict__ out) {
  __shared__ float red[256];
  const int t = blockIdx.x, tid = threadIdx.x;
  const float* row = x + (size_t)t * DMODEL;
  float s = 0.0f;
  for (int d = tid; d < DMODEL; d += 256) s += row[d];
  red[tid] = s; __syncthreads();
  for (int st = 128; st > 0; st >>= 1) {
    if (tid < st) red[tid] += red[tid + st];
    __syncthreads();
  }
  const float mu = red[0] / (float)DMODEL;
  __syncthreads();
  float s2 = 0.0f;
  for (int d = tid; d < DMODEL; d += 256) {
    const float dv = row[d] - mu;
    s2 += dv * dv;
  }
  red[tid] = s2; __syncthreads();
  for (int st = 128; st > 0; st >>= 1) {
    if (tid < st) red[tid] += red[tid + st];
    __syncthreads();
  }
  const float inv = rsqrtf(red[0] / (float)DMODEL + 1e-5f);
  for (int d = tid; d < DMODEL; d += 256)
    out[(size_t)t * DMODEL + d] = (row[d] - mu) * inv * g[d] + bta[d];
}

// Gate logits + top-2 softmax -> dense per-expert combine weights we[T,E].
__global__ void gate_kernel(const float* __restrict__ x, const float* __restrict__ gw,
                            const float* __restrict__ gb, float* __restrict__ we) {
  __shared__ float red[256];
  __shared__ float logits[NEXP];
  const int t = blockIdx.x, tid = threadIdx.x;
  const float* row = x + (size_t)t * DMODEL;
  for (int e = 0; e < NEXP; ++e) {
    float s = 0.0f;
    for (int d = tid; d < DMODEL; d += 256) s += row[d] * gw[(size_t)d * NEXP + e];
    red[tid] = s; __syncthreads();
    for (int st = 128; st > 0; st >>= 1) {
      if (tid < st) red[tid] += red[tid + st];
      __syncthreads();
    }
    if (tid == 0) logits[e] = red[0] + gb[e];
    __syncthreads();
  }
  if (tid == 0) {
    int i0 = 0; float v0 = logits[0];
    for (int e = 1; e < NEXP; ++e)
      if (logits[e] > v0) { v0 = logits[e]; i0 = e; }
    int i1 = -1; float v1 = -1e30f;
    for (int e = 0; e < NEXP; ++e)
      if (e != i0 && logits[e] > v1) { v1 = logits[e]; i1 = e; }
    float w1 = __expf(v1 - v0);
    const float sum = 1.0f + w1;
    const float w0 = 1.0f / sum;
    w1 /= sum;
    for (int e = 0; e < NEXP; ++e)
      we[(size_t)t * NEXP + e] = (e == i0) ? w0 : ((e == i1) ? w1 : 0.0f);
  }
}

__global__ void zero_kernel(float* __restrict__ p, int n) {
  const int i = blockIdx.x * 256 + threadIdx.x;
  if (i < n) p[i] = 0.0f;
}

// ---------------------------------------------------------------------------
extern "C" void kernel_launch(void* const* d_in, const int* in_sizes, int n_in,
                              void* d_out, int out_size, void* d_ws, size_t ws_size,
                              hipStream_t stream) {
  (void)in_sizes; (void)n_in; (void)out_size; (void)ws_size;
  const int*   src  = (const int*)  d_in[0];
  const float* emb  = (const float*)d_in[1];
  const float* qw   = (const float*)d_in[2];
  const float* qbi  = (const float*)d_in[3];
  const float* kw   = (const float*)d_in[4];
  const float* kbi  = (const float*)d_in[5];
  const float* vw   = (const float*)d_in[6];
  const float* vbi  = (const float*)d_in[7];
  const float* ow   = (const float*)d_in[8];
  const float* obi  = (const float*)d_in[9];
  const float* f1w  = (const float*)d_in[10];
  const float* f1b  = (const float*)d_in[11];
  const float* f2w  = (const float*)d_in[12];
  const float* f2b  = (const float*)d_in[13];
  const float* ln1g = (const float*)d_in[14];
  const float* ln1b = (const float*)d_in[15];
  const float* ln2g = (const float*)d_in[16];
  const float* ln2b = (const float*)d_in[17];
  const float* gw   = (const float*)d_in[18];
  const float* gb   = (const float*)d_in[19];
  const float* ew1  = (const float*)d_in[20];
  const float* eb1  = (const float*)d_in[21];
  const float* ew2  = (const float*)d_in[22];
  const float* eb2  = (const float*)d_in[23];
  const float* flng = (const float*)d_in[24];
  const float* flnb = (const float*)d_in[25];
  const float* outw = (const float*)d_in[26];
  const float* outb = (const float*)d_in[27];

  const size_t TD = (size_t)T_TOK * DMODEL;
  float* ws   = (float*)d_ws;
  float* x    = ws;                         // TD
  float* bufq = x + TD;                     // TD
  float* bufk = bufq + TD;                  // TD
  float* bufv = bufk + TD;                  // TD
  float* ao   = bufv + TD;                  // TD
  float* tmp  = ao + TD;                    // TD
  float* ff   = tmp + TD;                   // T*DFF
  float* moe  = ff + (size_t)T_TOK * DFF_;  // TD
  float* yb   = moe + TD;                   // TD
  float* cosb = yb + TD;                    // SEQ*32
  float* sinb = cosb + (size_t)SEQ * 32;    // SEQ*32
  float* we   = sinb + (size_t)SEQ * 32;    // T*E
  __bf16* qh  = (__bf16*)(we + (size_t)T_TOK * NEXP);  // TD bf16
  __bf16* kh  = qh + TD;                                // TD bf16
  __bf16* vt  = kh + TD;                                // TD bf16

  const dim3 blk(256);
  const dim3 g_dd(DMODEL / 64, T_TOK / 128);
  const dim3 g_dff(DFF_ / 64, T_TOK / 128);
  const dim3 g_out(VOCAB / 64, T_TOK / 128);
  const dim3 g_attn(SEQ / 64, NHEAD, BATCH);
  const dim3 ablk(128);

  yarn_kernel<<<SEQ, 32, 0, stream>>>(cosb, sinb);
  embed_kernel<<<(T_TOK * DMODEL + 255) / 256, 256, 0, stream>>>(src, emb, x);

  for (int l = 0; l < NLAYER; ++l) {
    const size_t wofs = (size_t)l * DMODEL * DMODEL;
    gemm_kernel<<<g_dd, blk, 0, stream>>>(x, qw + wofs, qbi + l * DMODEL,
                                          nullptr, nullptr, 0, bufq,
                                          T_TOK, DMODEL, DMODEL, 0, 0);
    gemm_kernel<<<g_dd, blk, 0, stream>>>(x, kw + wofs, kbi + l * DMODEL,
                                          nullptr, nullptr, 0, bufk,
                                          T_TOK, DMODEL, DMODEL, 0, 0);
    gemm_kernel<<<g_dd, blk, 0, stream>>>(x, vw + wofs, vbi + l * DMODEL,
                                          nullptr, nullptr, 0, bufv,
                                          T_TOK, DMODEL, DMODEL, 0, 0);
    rope_kernel<<<(T_TOK * NHEAD * 32 + 255) / 256, 256, 0, stream>>>(
        bufq, bufk, cosb, sinb);
    cvt_qkv_kernel<<<(T_TOK * DMODEL + 255) / 256, 256, 0, stream>>>(
        bufq, bufk, bufv, qh, kh, vt);
    attn_kernel<<<g_attn, ablk, 0, stream>>>(qh, kh, vt, ao);
    gemm_kernel<<<g_dd, blk, 0, stream>>>(ao, ow + wofs, obi + l * DMODEL,
                                          x, nullptr, 0, tmp,
                                          T_TOK, DMODEL, DMODEL, 0, 0);
    ln_kernel<<<T_TOK, 256, 0, stream>>>(tmp, ln1g + l * DMODEL,
                                         ln1b + l * DMODEL, x);
    gemm_kernel<<<g_dff, blk, 0, stream>>>(x, f1w + (size_t)l * DMODEL * DFF_,
                                           f1b + l * DFF_, nullptr, nullptr, 0,
                                           ff, T_TOK, DMODEL, DFF_, 1, 0);
    gemm_kernel<<<g_dd, blk, 0, stream>>>(ff, f2w + (size_t)l * DFF_ * DMODEL,
                                          f2b + l * DMODEL, x, nullptr, 0, tmp,
                                          T_TOK, DFF_, DMODEL, 0, 0);
    ln_kernel<<<T_TOK, 256, 0, stream>>>(tmp, ln2g + l * DMODEL,
                                         ln2b + l * DMODEL, x);
  }

  // MoE (dense over all experts, matching the reference)
  gate_kernel<<<T_TOK, 256, 0, stream>>>(x, gw, gb, we);
  zero_kernel<<<((int)TD + 255) / 256, 256, 0, stream>>>(moe, (int)TD);
  for (int e = 0; e < NEXP; ++e) {
    gemm_kernel<<<g_dff, blk, 0, stream>>>(x, ew1 + (size_t)e * DMODEL * DFF_,
                                           eb1 + (size_t)e * DFF_, nullptr,
                                           nullptr, 0, ff,
                                           T_TOK, DMODEL, DFF_, 1, 0);
    gemm_kernel<<<g_dd, blk, 0, stream>>>(ff, ew2 + (size_t)e * DFF_ * DMODEL,
                                          eb2 + (size_t)e * DMODEL, nullptr,
                                          we + e, NEXP, moe,
                                          T_TOK, DFF_, DMODEL, 0, 1);
  }
  ln_kernel<<<T_TOK, 256, 0, stream>>>(moe, flng, flnb, yb);
  gemm_kernel<<<g_out, blk, 0, stream>>>(yb, outw, outb, nullptr, nullptr, 0,
                                         (float*)d_out, T_TOK, DMODEL, VOCAB,
                                         0, 0);
}
